// FreqProcessingUnit_66632122630637
// MI455X (gfx1250) — matrix operand order repaired and checked by reading, not compiled, not run on hardware
//
#include <hip/hip_runtime.h>
#include <hip/hip_bf16.h>

typedef _Float16 f16;
typedef __attribute__((ext_vector_type(16))) _Float16 v16h;
typedef __attribute__((ext_vector_type(8)))  float    v8f;

union H16 { v16h v; f16 h[16]; uint4 q[2]; };
union F8  { v8f  v; float f[8]; };

#define N_PIX 2304
#define C_DIM 128
#define NHEAD 4
#define HC 32
#define KCLU 16

// ---------------- fragment loaders (wave32, CDNA5 WMMA layouts) ----------------
// A-matrix fragment (16 rows x 32 K, f16), source tile row-major with rowStride halves.
// Contiguous k order used here: lane row = lane&15; K(e) = e + 16*(lane>>4)  -> two 16B loads.
__device__ __forceinline__ v16h loadA32(const f16* base, int rowStride, int lane) {
  H16 x;
  int row = lane & 15, qq = lane >> 4;
  const f16* p = base + row * rowStride;
  x.q[0] = *(const uint4*)(p + 16 * qq);
  x.q[1] = *(const uint4*)(p + 16 * qq + 8);
  return x.v;
}
// B-matrix fragment (32 K x 16 N, f16) from a tile stored as T[n][k] row-major
// (i.e. B = T^T).  Same contiguous k order as A: lane col n = lane&15; K(e) = e + 16*(lane>>4)
// -> 16 contiguous halves = two 16B loads.
__device__ __forceinline__ v16h loadB32(const f16* base, int rowStride, int lane) {
  H16 x;
  int row = lane & 15, qq = lane >> 4;
  const uint4* p = (const uint4*)(base + row * rowStride + 16 * qq);
  x.q[0] = p[0];
  x.q[1] = p[1];
  return x.v;
}
__device__ __forceinline__ v8f wmma32(v16h a, v16h b, v8f c) {
  return __builtin_amdgcn_wmma_f32_16x16x32_f16(false, a, false, b, (short)0, c, false, false);
}

// ---------------- prep: f16 conversions + transposes ----------------
__global__ void prep_kernel(const float* __restrict__ q_img, const float* __restrict__ k_img,
                            const float* __restrict__ v_img,
                            const float* __restrict__ wq, const float* __restrict__ wk,
                            const float* __restrict__ wv,
                            const float* __restrict__ w1a, const float* __restrict__ w1b,
                            const float* __restrict__ w2a, const float* __restrict__ w2b,
                            f16* __restrict__ xq, f16* __restrict__ xk, f16* __restrict__ xv,
                            f16* __restrict__ wqT, f16* __restrict__ wkT, f16* __restrict__ wvT,
                            f16* __restrict__ w1aT, f16* __restrict__ w1bT,
                            f16* __restrict__ w2aT, f16* __restrict__ w2bT) {
  int task = blockIdx.y;
  int idx = blockIdx.x * 256 + threadIdx.x;
  if (task < 3) {                       // images [C][N] -> f16 [N][C]
    const float* src = task == 0 ? q_img : (task == 1 ? k_img : v_img);
    f16* dst = task == 0 ? xq : (task == 1 ? xk : xv);
    if (idx < N_PIX * C_DIM) {
      int p = idx >> 7, c = idx & 127;
      dst[idx] = (f16)src[c * N_PIX + p];
    }
  } else if (task < 6) {                // [128 in][128 out] -> T [out][in]
    const float* src = task == 3 ? wq : (task == 4 ? wk : wv);
    f16* dst = task == 3 ? wqT : (task == 4 ? wkT : wvT);
    if (idx < 128 * 128) {
      int o = idx >> 7, i = idx & 127;
      dst[idx] = (f16)src[i * 128 + o];
    }
  } else if (task == 6 || task == 8) {  // [128 in][256 out] -> T [256][128]
    const float* src = task == 6 ? w1a : w2a;
    f16* dst = task == 6 ? w1aT : w2aT;
    if (idx < 256 * 128) {
      int o = idx >> 7, i = idx & 127;
      dst[idx] = (f16)src[i * 256 + o];
    }
  } else {                              // [256 in][128 out] -> T [128][256]
    const float* src = task == 7 ? w1b : w2b;
    f16* dst = task == 7 ? w1bT : w2bT;
    if (idx < 128 * 256) {
      int o = idx >> 8, i = idx & 255;
      dst[idx] = (f16)src[i * 128 + o];
    }
  }
}

// ---------------- projections: Y = X @ W + b via WMMA ----------------
__global__ void proj_kernel(const f16* __restrict__ xq, const f16* __restrict__ xk,
                            const f16* __restrict__ xv,
                            const f16* __restrict__ wqT, const f16* __restrict__ wkT,
                            const f16* __restrict__ wvT,
                            const float* __restrict__ bq, const float* __restrict__ bk,
                            const float* __restrict__ bv,
                            f16* __restrict__ qf16, f16* __restrict__ kf16,
                            f16* __restrict__ vT, float* __restrict__ vf32) {
  int lane = threadIdx.x;
  int qb = blockIdx.x * 16, nb = blockIdx.y * 16, z = blockIdx.z;
  const f16* X = z == 0 ? xq : (z == 1 ? xk : xv);
  const f16* WT = z == 0 ? wqT : (z == 1 ? wkT : wvT);
  const float* bias = z == 0 ? bq : (z == 1 ? bk : bv);
  F8 acc;
  for (int r = 0; r < 8; ++r) acc.f[r] = 0.f;
  for (int kc = 0; kc < 4; ++kc) {
    v16h a = loadA32(X + qb * C_DIM + kc * 32, C_DIM, lane);
    v16h b = loadB32(WT + nb * C_DIM + kc * 32, C_DIM, lane);
    acc.v = wmma32(a, b, acc.v);
  }
  int nc = lane & 15, half = lane >> 4;
  int col = nb + nc;
  float bb = bias[col];
  for (int r = 0; r < 8; ++r) {
    int p = qb + r + 8 * half;
    float y = acc.f[r] + bb;
    if (z == 0) qf16[p * C_DIM + col] = (f16)y;
    else if (z == 1) kf16[p * C_DIM + col] = (f16)y;
    else { vf32[p * C_DIM + col] = y; vT[col * N_PIX + p] = (f16)y; }
  }
}

// ---------------- cluster centers from K ----------------
__global__ void centers_kernel(const f16* __restrict__ kf, const int* __restrict__ labels,
                               f16* __restrict__ cf) {
  int k = blockIdx.x, c = threadIdx.x;
  float s = 0.f, cnt = 0.f;
  for (int p = 0; p < N_PIX; ++p) {
    int lab = labels[p];
    float val = (float)kf[p * C_DIM + c];
    bool sel = (lab == k);
    s += sel ? val : 0.f;
    cnt += sel ? 1.f : 0.f;
  }
  cf[k * C_DIM + c] = (f16)(s / (cnt + 1e-6f));
}

// ---------------- fused flash attention with cluster fixup ----------------
__global__ void attn_kernel(const f16* __restrict__ qf, const f16* __restrict__ kf,
                            const f16* __restrict__ vT, const f16* __restrict__ cf,
                            const float* __restrict__ pc, const int* __restrict__ labels,
                            f16* __restrict__ aout) {
  __shared__ float acS[16 * 16];          // Q @ centers^T tile
  __shared__ int labI[16];
  __shared__ f16 Pt[4][16 * 32];          // per-wave P transpose staging
  __shared__ float mS[4][16], lS[4][16], Ow[4][16 * 32];
  int tid = threadIdx.x, lane = tid & 31, w = tid >> 5;
  int qb = blockIdx.x * 16, h = blockIdx.y;
  int half = lane >> 4, nc = lane & 15;

  v16h aQ = loadA32(qf + qb * C_DIM + h * HC, C_DIM, lane);

  if (w == 0) {
    v16h bC = loadB32(cf + h * HC, C_DIM, lane);
    F8 ac;
    for (int r = 0; r < 8; ++r) ac.f[r] = 0.f;
    ac.v = wmma32(aQ, bC, ac.v);
    for (int r = 0; r < 8; ++r) acS[(r + 8 * half) * 16 + nc] = ac.f[r];
  }
  if (tid < 16) labI[tid] = labels[qb + tid];
  __syncthreads();

  const float invS = 0.17677669529663687f;  // 1/sqrt(32)
  float m[8], l[8];
  F8 o0, o1;
  for (int r = 0; r < 8; ++r) { m[r] = -1e30f; l[r] = 0.f; o0.f[r] = 0.f; o1.f[r] = 0.f; }

  for (int c = w; c < N_PIX / 32; c += 4) {
    int jb = c * 32;
    int labj = labels[jb + lane];
    v16h bK0 = loadB32(kf + jb * C_DIM + h * HC, C_DIM, lane);
    v16h bK1 = loadB32(kf + (jb + 16) * C_DIM + h * HC, C_DIM, lane);
    F8 s0, s1;
    for (int r = 0; r < 8; ++r) { s0.f[r] = 0.f; s1.f[r] = 0.f; }
    s0.v = wmma32(aQ, bK0, s0.v);
    s1.v = wmma32(aQ, bK1, s1.v);
    int lj0 = __shfl(labj, nc, 32);
    int lj1 = __shfl(labj, 16 + nc, 32);
    float p0[8], p1[8];
    for (int r = 0; r < 8; ++r) {
      int i = r + 8 * half;
      int li = labI[i];
      const float* pcRow = pc + (qb + i) * N_PIX + jb;
      float v0 = (li == lj0) ? s0.f[r] : acS[i * 16 + lj0] * pcRow[nc];
      float v1 = (li == lj1) ? s1.f[r] : acS[i * 16 + lj1] * pcRow[16 + nc];
      v0 *= invS; v1 *= invS;
      float cm = fmaxf(v0, v1);
      for (int msk = 1; msk < 16; msk <<= 1) cm = fmaxf(cm, __shfl_xor(cm, msk, 32));
      float mn = fmaxf(m[r], cm);
      float sc = __expf(m[r] - mn);
      float e0 = __expf(v0 - mn), e1 = __expf(v1 - mn);
      float lr = e0 + e1;
      for (int msk = 1; msk < 16; msk <<= 1) lr += __shfl_xor(lr, msk, 32);
      m[r] = mn; l[r] = l[r] * sc + lr;
      o0.f[r] *= sc; o1.f[r] *= sc;
      p0[r] = e0; p1[r] = e1;
    }
    // transpose P through per-wave LDS (C-layout -> A-layout)
    f16* pt = Pt[w];
    for (int r = 0; r < 8; ++r) {
      int i = r + 8 * half;
      pt[i * 32 + nc] = (f16)p0[r];
      pt[i * 32 + 16 + nc] = (f16)p1[r];
    }
    asm volatile("s_wait_dscnt 0" ::: "memory");
    v16h aP = loadA32(pt, 32, lane);
    v16h bV0 = loadB32(vT + (h * HC) * N_PIX + jb, N_PIX, lane);
    v16h bV1 = loadB32(vT + (h * HC + 16) * N_PIX + jb, N_PIX, lane);
    o0.v = wmma32(aP, bV0, o0.v);
    o1.v = wmma32(aP, bV1, o1.v);
  }
  if (nc == 0)
    for (int r = 0; r < 8; ++r) { mS[w][r + 8 * half] = m[r]; lS[w][r + 8 * half] = l[r]; }
  for (int r = 0; r < 8; ++r) {
    int i = r + 8 * half;
    Ow[w][i * 32 + nc] = o0.f[r];
    Ow[w][i * 32 + 16 + nc] = o1.f[r];
  }
  __syncthreads();
  for (int e = tid; e < 512; e += 128) {
    int i = e >> 5, d = e & 31;
    float M = fmaxf(fmaxf(mS[0][i], mS[1][i]), fmaxf(mS[2][i], mS[3][i]));
    float num = 0.f, den = 0.f;
    for (int ww = 0; ww < 4; ++ww) {
      float f = __expf(mS[ww][i] - M);
      num += f * Ow[ww][i * 32 + d];
      den += f * lS[ww][i];
    }
    aout[(qb + i) * C_DIM + h * HC + d] = (f16)(num / den);
  }
}

// ---------------- fused MLP1 + residual(v) ----------------
__global__ void mlp1_kernel(const f16* __restrict__ ain, const f16* __restrict__ w1aT,
                            const float* __restrict__ b1a, const f16* __restrict__ w1bT,
                            const float* __restrict__ b1b, const float* __restrict__ vf32,
                            float* __restrict__ rs1f, f16* __restrict__ rs1h) {
  __shared__ f16 ht[16 * 256];
  int tid = threadIdx.x, lane = tid & 31, w = tid >> 5;
  int qb = blockIdx.x * 16;
  int half = lane >> 4, nc = lane & 15;
  v16h aA[4];
  for (int kc = 0; kc < 4; ++kc) aA[kc] = loadA32(ain + qb * C_DIM + kc * 32, C_DIM, lane);
  for (int t = 0; t < 4; ++t) {
    int nb = w + 4 * t;  // 0..15 over 256 hidden
    F8 acc;
    for (int r = 0; r < 8; ++r) acc.f[r] = 0.f;
    for (int kc = 0; kc < 4; ++kc) {
      v16h b = loadB32(w1aT + (nb * 16) * C_DIM + kc * 32, C_DIM, lane);
      acc.v = wmma32(aA[kc], b, acc.v);
    }
    int col = nb * 16 + nc;
    float bb = b1a[col];
    for (int r = 0; r < 8; ++r) {
      float hv = acc.f[r] + bb;
      hv = hv > 0.f ? hv : 0.01f * hv;
      ht[(r + 8 * half) * 256 + col] = (f16)hv;
    }
  }
  __syncthreads();
  for (int t = 0; t < 2; ++t) {
    int nb = w + 4 * t;  // 0..7 over 128 out
    F8 acc;
    for (int r = 0; r < 8; ++r) acc.f[r] = 0.f;
    for (int kc = 0; kc < 8; ++kc) {
      v16h a = loadA32(ht + kc * 32, 256, lane);
      v16h b = loadB32(w1bT + (nb * 16) * 256 + kc * 32, 256, lane);
      acc.v = wmma32(a, b, acc.v);
    }
    int col = nb * 16 + nc;
    float bb = b1b[col];
    for (int r = 0; r < 8; ++r) {
      int p = qb + r + 8 * half;
      float y = acc.f[r] + bb + vf32[p * C_DIM + col];
      rs1f[p * C_DIM + col] = y;
      rs1h[p * C_DIM + col] = (f16)y;
    }
  }
}

// ---------------- fused MLP2 + residual(rs1), final [C][H][W] output ----------------
__global__ void mlp2_kernel(const f16* __restrict__ rs1h, const f16* __restrict__ w2aT,
                            const float* __restrict__ b2a, const f16* __restrict__ w2bT,
                            const float* __restrict__ b2b, const float* __restrict__ rs1f,
                            float* __restrict__ dout) {
  __shared__ f16 ht[16 * 256];
  int tid = threadIdx.x, lane = tid & 31, w = tid >> 5;
  int qb = blockIdx.x * 16;
  int half = lane >> 4, nc = lane & 15;
  v16h aA[4];
  for (int kc = 0; kc < 4; ++kc) aA[kc] = loadA32(rs1h + qb * C_DIM + kc * 32, C_DIM, lane);
  for (int t = 0; t < 4; ++t) {
    int nb = w + 4 * t;
    F8 acc;
    for (int r = 0; r < 8; ++r) acc.f[r] = 0.f;
    for (int kc = 0; kc < 4; ++kc) {
      v16h b = loadB32(w2aT + (nb * 16) * C_DIM + kc * 32, C_DIM, lane);
      acc.v = wmma32(aA[kc], b, acc.v);
    }
    int col = nb * 16 + nc;
    float bb = b2a[col];
    for (int r = 0; r < 8; ++r) {
      float hv = acc.f[r] + bb;
      hv = hv > 0.f ? hv : 0.01f * hv;
      ht[(r + 8 * half) * 256 + col] = (f16)hv;
    }
  }
  __syncthreads();
  for (int t = 0; t < 2; ++t) {
    int nb = w + 4 * t;
    F8 acc;
    for (int r = 0; r < 8; ++r) acc.f[r] = 0.f;
    for (int kc = 0; kc < 8; ++kc) {
      v16h a = loadA32(ht + kc * 32, 256, lane);
      v16h b = loadB32(w2bT + (nb * 16) * 256 + kc * 32, 256, lane);
      acc.v = wmma32(a, b, acc.v);
    }
    int col = nb * 16 + nc;
    float bb = b2b[col];
    for (int r = 0; r < 8; ++r) {
      int p = qb + r + 8 * half;
      float y = acc.f[r] + bb + rs1f[p * C_DIM + col];
      dout[col * N_PIX + p] = y;  // [C][H][W]
    }
  }
}

// ---------------- workspace layout (bytes, 256-aligned) ----------------
#define WS_XQ     0u
#define WS_XK     589824u
#define WS_XV     1179648u
#define WS_WQT    1769472u
#define WS_WKT    1802240u
#define WS_WVT    1835008u
#define WS_W1AT   1867776u
#define WS_W1BT   1933312u
#define WS_W2AT   1998848u
#define WS_W2BT   2064384u
#define WS_QF16   2129920u
#define WS_KF16   2719744u
#define WS_VT16   3309568u
#define WS_VF32   3899392u
#define WS_CF16   5079040u
#define WS_AOF16  5083136u
#define WS_RS1F   5672960u
#define WS_RS1H   6852608u

extern "C" void kernel_launch(void* const* d_in, const int* in_sizes, int n_in,
                              void* d_out, int out_size, void* d_ws, size_t ws_size,
                              hipStream_t stream) {
  (void)in_sizes; (void)n_in; (void)out_size; (void)ws_size;
  const float* q_img = (const float*)d_in[0];
  const float* k_img = (const float*)d_in[1];
  const float* v_img = (const float*)d_in[2];
  const float* pc    = (const float*)d_in[3];
  const int*   labels= (const int*)d_in[4];
  const float* wq = (const float*)d_in[5];  const float* bq = (const float*)d_in[6];
  const float* wk = (const float*)d_in[7];  const float* bk = (const float*)d_in[8];
  const float* wv = (const float*)d_in[9];  const float* bv = (const float*)d_in[10];
  const float* w1a = (const float*)d_in[11]; const float* b1a = (const float*)d_in[12];
  const float* w1b = (const float*)d_in[13]; const float* b1b = (const float*)d_in[14];
  const float* w2a = (const float*)d_in[15]; const float* b2a = (const float*)d_in[16];
  const float* w2b = (const float*)d_in[17]; const float* b2b = (const float*)d_in[18];
  char* ws = (char*)d_ws;
  f16* xq   = (f16*)(ws + WS_XQ);   f16* xk   = (f16*)(ws + WS_XK);   f16* xv  = (f16*)(ws + WS_XV);
  f16* wqT  = (f16*)(ws + WS_WQT);  f16* wkT  = (f16*)(ws + WS_WKT);  f16* wvT = (f16*)(ws + WS_WVT);
  f16* w1aT = (f16*)(ws + WS_W1AT); f16* w1bT = (f16*)(ws + WS_W1BT);
  f16* w2aT = (f16*)(ws + WS_W2AT); f16* w2bT = (f16*)(ws + WS_W2BT);
  f16* qf16 = (f16*)(ws + WS_QF16); f16* kf16 = (f16*)(ws + WS_KF16);
  f16* vT   = (f16*)(ws + WS_VT16); float* vf32 = (float*)(ws + WS_VF32);
  f16* cf16 = (f16*)(ws + WS_CF16); f16* aout = (f16*)(ws + WS_AOF16);
  float* rs1f = (float*)(ws + WS_RS1F); f16* rs1h = (f16*)(ws + WS_RS1H);
  float* out = (float*)d_out;

  prep_kernel<<<dim3(1152, 9), 256, 0, stream>>>(q_img, k_img, v_img, wq, wk, wv,
                                                 w1a, w1b, w2a, w2b,
                                                 xq, xk, xv, wqT, wkT, wvT,
                                                 w1aT, w1bT, w2aT, w2bT);
  proj_kernel<<<dim3(144, 8, 3), 32, 0, stream>>>(xq, xk, xv, wqT, wkT, wvT,
                                                  bq, bk, bv, qf16, kf16, vT, vf32);
  centers_kernel<<<16, 128, 0, stream>>>(kf16, labels, cf16);
  attn_kernel<<<dim3(144, 4), 128, 0, stream>>>(qf16, kf16, vT, cf16, pc, labels, aout);
  mlp1_kernel<<<144, 128, 0, stream>>>(aout, w1aT, b1a, w1bT, b1b, vf32, rs1f, rs1h);
  mlp2_kernel<<<144, 128, 0, stream>>>(rs1h, w2aT, b2a, w2bT, b2b, rs1f, out);
}
